// BertModelWrapper_noise_6906307412329
// MI455X (gfx1250) — matrix operand order repaired and checked by reading, not compiled, not run on hardware
//
#include <hip/hip_runtime.h>
#include <hip/hip_bf16.h>

#define B_  16
#define S_  512
#define H_  768
#define L_  12
#define NH_ 12
#define FF_ 3072
#define DH_ 64

typedef __attribute__((ext_vector_type(16))) __bf16 v16bf;
typedef __attribute__((ext_vector_type(8)))  __bf16 v8bf;
typedef __attribute__((ext_vector_type(4)))  __bf16 v4bf;
typedef __attribute__((ext_vector_type(2)))  __bf16 v2bf;
typedef __attribute__((ext_vector_type(8)))  float  v8f;

// ---------------------------------------------------------------------------
// WMMA helper: D = A(16x32 bf16) * B(32x16 bf16) + C(16x16 f32)
// ---------------------------------------------------------------------------
__device__ inline v8f wmma_bf16(v16bf a, v16bf b, v8f c) {
  return __builtin_amdgcn_wmma_f32_16x16x32_bf16(
      /*neg_a=*/false, a, /*neg_b=*/false, b,
      /*c_mod=*/(short)0, c, /*reuse_a=*/false, /*reuse_b=*/false);
}

// A-matrix fragment (16x32 bf16) from LDS tile [M][K], in this file's contiguous k order (B uses the same):
//   lane: m = lane&15, laneHalf = lane>>4
//   element j<8 : K = kCol + laneHalf*16 + j        (contiguous 16B)
//   element j>=8: K = kCol + laneHalf*16 + j         (contiguous 16B)
__device__ inline v16bf load_a_frag(const __bf16* __restrict__ smem, int lda,
                                    int mRow, int kCol) {
  const int lane = threadIdx.x & 31;
  const int m = lane & 15;
  const int lh = lane >> 4;
  const __bf16* p = smem + (mRow + m) * lda + kCol + lh * 16;
  v16bf a;
#pragma unroll
  for (int j = 0; j < 8; ++j) a[j] = p[j];
#pragma unroll
  for (int j = 0; j < 8; ++j) a[8 + j] = p[8 + j];
  return a;
}

// B-matrix fragment (32x16 bf16) from a TRANSPOSED LDS tile [N][K]:
//   lane: n = lane&15, laneHalf = lane>>4; element j: K = kRow + laneHalf*16+j
//   -> 16 contiguous bf16 (32B) per lane => ds_load_b128 x2
__device__ inline v16bf load_b_frag_t(const __bf16* __restrict__ smem, int ldt,
                                      int kRow, int nCol) {
  const int lane = threadIdx.x & 31;
  const int n = lane & 15;
  const int lh = lane >> 4;
  const __bf16* p = smem + (nCol + n) * ldt + kRow + lh * 16;
  v16bf b;
#pragma unroll
  for (int j = 0; j < 16; ++j) b[j] = p[j];
  return b;
}

__device__ inline void zero_acc(v8f& a) {
#pragma unroll
  for (int r = 0; r < 8; ++r) a[r] = 0.0f;
}

__device__ inline v4bf cvt4(float x, float y, float z, float w) {
  v4bf o;
  o[0] = (__bf16)x; o[1] = (__bf16)y; o[2] = (__bf16)z; o[3] = (__bf16)w;
  return o;
}

// Async bf16 tile copy HBM -> LDS (raw bytes, ASYNCcnt-tracked).
__device__ inline void async_copy_b128_to_lds(unsigned lds_addr,
                                              const void* gptr) {
  asm volatile("global_load_async_to_lds_b128 %0, %1, off"
               :
               : "v"(lds_addr), "v"((unsigned long long)gptr)
               : "memory");
}
__device__ inline void wait_asynccnt0() {
  asm volatile("s_wait_asynccnt 0x0" ::: "memory");
}

// ---------------------------------------------------------------------------
// Generic GEMM: C[M,N] = A[M,K] @ W[K,N] + bias[N]  (fp32 in HBM, bf16 staged
// in LDS, WMMA f32 accumulate). Block 256 thr / 8 waves; tile 128x128, K-step
// 32; wave tile 64x32. Requires M%128==0, N%128==0, K%32==0.
// ---------------------------------------------------------------------------
__global__ __launch_bounds__(256) void gemm_bf16_kernel(
    float* __restrict__ C, const float* __restrict__ A,
    const float* __restrict__ W, const float* __restrict__ bias,
    int M, int N, int K) {
  __shared__ __bf16 As[128 * 40];  // [m][k], +8 pad keeps 16B rows
  __shared__ __bf16 Bt[128 * 40];  // [n][k] transposed weight tile
  const int tid = threadIdx.x;
  const int wid = tid >> 5;
  const int lane = tid & 31;
  const int m0 = blockIdx.x * 128;
  const int n0 = blockIdx.y * 128;
  const int wm = wid & 1;   // 64-row half
  const int wn = wid >> 1;  // 32-col quarter

  v8f acc[4][2];
#pragma unroll
  for (int mi = 0; mi < 4; ++mi)
#pragma unroll
    for (int ni = 0; ni < 2; ++ni) zero_acc(acc[mi][ni]);

  for (int k0 = 0; k0 < K; k0 += 32) {
    // Prefetch next K-tile while we stage the current one
    const int nk = k0 + 32;
    if (nk < K) {
      __builtin_prefetch(&A[(size_t)(m0 + (tid >> 1)) * K + nk + (tid & 1) * 16], 0, 3);
      __builtin_prefetch(&W[(size_t)(nk + (tid >> 4)) * N + n0 + ((tid & 15) << 3)], 0, 3);
    }
    __syncthreads();
    // Stage A tile 128x32: float4 loads, packed v4bf stores
#pragma unroll
    for (int i = 0; i < 4; ++i) {
      const int idx = i * 256 + tid;
      const int row = idx >> 3, c = (idx & 7) * 4;
      const float4 f =
          *(const float4*)(A + (size_t)(m0 + row) * K + k0 + c);
      *(v4bf*)(As + row * 40 + c) = cvt4(f.x, f.y, f.z, f.w);
    }
    // Stage W tile 32x128 transposed -> Bt[n][k]; 4 coalesced b32 streams
#pragma unroll
    for (int i = 0; i < 4; ++i) {
      const int idx = i * 256 + tid;
      const int n = idx & 127, c = (idx >> 7) * 4;
      const size_t base = (size_t)(k0 + c) * N + n0 + n;
      *(v4bf*)(Bt + n * 40 + c) =
          cvt4(W[base], W[base + N], W[base + 2 * N], W[base + 3 * N]);
    }
    __syncthreads();

    v16bf bf[2];
#pragma unroll
    for (int ni = 0; ni < 2; ++ni)
      bf[ni] = load_b_frag_t(Bt, 40, 0, wn * 32 + ni * 16);
#pragma unroll
    for (int mi = 0; mi < 4; ++mi) {
      v16bf af = load_a_frag(As, 40, wm * 64 + mi * 16, 0);
#pragma unroll
      for (int ni = 0; ni < 2; ++ni)
        acc[mi][ni] = wmma_bf16(af, bf[ni], acc[mi][ni]);
    }
  }

  // Epilogue: + bias, store fp32. C layout: N=lane&15, M=r+(lane>>4)*8
  const int lh = lane >> 4;
  const int nn = lane & 15;
#pragma unroll
  for (int mi = 0; mi < 4; ++mi)
#pragma unroll
    for (int ni = 0; ni < 2; ++ni) {
      const int ncol = n0 + wn * 32 + ni * 16 + nn;
      const float bvl = bias[ncol];
#pragma unroll
      for (int r = 0; r < 8; ++r) {
        const int mrow = m0 + wm * 64 + mi * 16 + r + lh * 8;
        C[(size_t)mrow * N + ncol] = acc[mi][ni][r] + bvl;
      }
    }
}

// ---------------------------------------------------------------------------
// Attention scores: scores[b,h,q,k] = (Q_h[q,:] . K_h[k,:]) / 8 + mask[b,k]
// Block: (bh, 64 q-rows, 128 keys). K-dim = 64 (2 WMMA steps).
// Ks staged [key][d] => coalesced fill AND contiguous B-frag reads.
// ---------------------------------------------------------------------------
__global__ __launch_bounds__(256) void attn_scores_kernel(
    float* __restrict__ scores, const float* __restrict__ q,
    const float* __restrict__ k, const float* __restrict__ amask) {
  __shared__ __bf16 Qs[64 * 72];    // [q][d] +8 pad
  __shared__ __bf16 Ks[128 * 72];   // [key][d] +8 pad
  const int tid = threadIdx.x;
  const int wid = tid >> 5;
  const int lane = tid & 31;
  const int bh = blockIdx.z;
  const int b = bh / NH_;
  const int h = bh % NH_;
  const int q0 = blockIdx.y * 64;
  const int k0 = blockIdx.x * 128;

#pragma unroll
  for (int i = 0; i < 4; ++i) {  // Q tile: 64x64 = 1024 float4
    const int idx = i * 256 + tid;
    const int row = idx >> 4, c = (idx & 15) * 4;
    const float4 f =
        *(const float4*)(q + (size_t)(b * S_ + q0 + row) * H_ + h * DH_ + c);
    *(v4bf*)(Qs + row * 72 + c) = cvt4(f.x, f.y, f.z, f.w);
  }
#pragma unroll
  for (int i = 0; i < 8; ++i) {  // K tile: 128x64 = 2048 float4
    const int idx = i * 256 + tid;
    const int row = idx >> 4, c = (idx & 15) * 4;
    const float4 f =
        *(const float4*)(k + (size_t)(b * S_ + k0 + row) * H_ + h * DH_ + c);
    *(v4bf*)(Ks + row * 72 + c) = cvt4(f.x, f.y, f.z, f.w);
  }
  __syncthreads();

  const int wm = wid & 1;
  const int wn = wid >> 1;
  v8f acc[2][2];
#pragma unroll
  for (int mi = 0; mi < 2; ++mi)
#pragma unroll
    for (int ni = 0; ni < 2; ++ni) zero_acc(acc[mi][ni]);

#pragma unroll
  for (int step = 0; step < 2; ++step) {
    const int kc = step * 32;
    v16bf bf[2];
#pragma unroll
    for (int ni = 0; ni < 2; ++ni)
      bf[ni] = load_b_frag_t(Ks, 72, kc, wn * 32 + ni * 16);
#pragma unroll
    for (int mi = 0; mi < 2; ++mi) {
      v16bf af = load_a_frag(Qs, 72, wm * 32 + mi * 16, kc);
#pragma unroll
      for (int ni = 0; ni < 2; ++ni)
        acc[mi][ni] = wmma_bf16(af, bf[ni], acc[mi][ni]);
    }
  }

  const int lh = lane >> 4;
  const int nn = lane & 15;
#pragma unroll
  for (int mi = 0; mi < 2; ++mi)
#pragma unroll
    for (int ni = 0; ni < 2; ++ni) {
      const int key = k0 + wn * 32 + ni * 16 + nn;
      const float mval = (1.0f - amask[b * S_ + key]) * -10000.0f;
#pragma unroll
      for (int r = 0; r < 8; ++r) {
        const int qq = q0 + wm * 32 + mi * 16 + r + lh * 8;
        scores[((size_t)bh * S_ + qq) * S_ + key] =
            acc[mi][ni][r] * 0.125f + mval;
      }
    }
}

// ---------------------------------------------------------------------------
// Row softmax over 512 keys; one wave per row; float2 in, packed bf16 out.
// ---------------------------------------------------------------------------
__global__ __launch_bounds__(128) void softmax_kernel(
    __bf16* __restrict__ probs, const float* __restrict__ scores) {
  const int row = blockIdx.x * 4 + (threadIdx.x >> 5);
  const int lane = threadIdx.x & 31;
  const float2* sp = (const float2*)(scores + (size_t)row * S_);
  float v[16];
  float mx = -3.0e38f;
#pragma unroll
  for (int i = 0; i < 8; ++i) {
    const float2 f = sp[i * 32 + lane];
    v[2 * i] = f.x;
    v[2 * i + 1] = f.y;
    mx = fmaxf(mx, fmaxf(f.x, f.y));
  }
#pragma unroll
  for (int off = 16; off > 0; off >>= 1) mx = fmaxf(mx, __shfl_xor(mx, off, 32));
  float sum = 0.0f;
#pragma unroll
  for (int i = 0; i < 16; ++i) {
    v[i] = __expf(v[i] - mx);
    sum += v[i];
  }
#pragma unroll
  for (int off = 16; off > 0; off >>= 1) sum += __shfl_xor(sum, off, 32);
  const float inv = 1.0f / sum;
  v2bf* pp = (v2bf*)(probs + (size_t)row * S_);
#pragma unroll
  for (int i = 0; i < 8; ++i) {
    v2bf o;
    o[0] = (__bf16)(v[2 * i] * inv);
    o[1] = (__bf16)(v[2 * i + 1] * inv);
    pp[i * 32 + lane] = o;
  }
}

// ---------------------------------------------------------------------------
// ctx[b,q,h*64+d] = sum_k probs[b,h,q,k] * v[b,k,h*64+d]
// Block: (bh, 128 q rows). Probs tile staged with ASYNC copies (raw bf16,
// ASYNCcnt); V tile staged transposed [d][key] for contiguous B-frag reads.
// ---------------------------------------------------------------------------
__global__ __launch_bounds__(256) void attn_ctx_kernel(
    float* __restrict__ ctx, const __bf16* __restrict__ probs,
    const float* __restrict__ v) {
  __shared__ __bf16 Ps[128 * 40];  // [q][key] +8 pad (rows 80B, 16B-aligned)
  __shared__ __bf16 Vt[64 * 40];   // [d][key] +8 pad
  const int tid = threadIdx.x;
  const int wid = tid >> 5;
  const int lane = tid & 31;
  const int bh = blockIdx.z;
  const int b = bh / NH_;
  const int h = bh % NH_;
  const int q0 = blockIdx.x * 128;

  v8f acc[4];
#pragma unroll
  for (int ni = 0; ni < 4; ++ni) zero_acc(acc[ni]);

  for (int k0 = 0; k0 < S_; k0 += 32) {
    __syncthreads();
    // Probs tile 128x32 bf16: async HBM->LDS (b128 per lane-op, 2 per thread)
#pragma unroll
    for (int i = 0; i < 2; ++i) {
      const int cidx = i * 256 + tid;
      const int row = cidx >> 2, c8 = (cidx & 3) * 8;
      const unsigned lds_addr = (unsigned)(unsigned long long)(&Ps[row * 40 + c8]);
      const __bf16* g = probs + ((size_t)bh * S_ + q0 + row) * S_ + k0 + c8;
      async_copy_b128_to_lds(lds_addr, (const void*)g);
    }
    // V tile 32x64 -> Vt[d][key]; 4 coalesced b32 streams, packed stores
#pragma unroll
    for (int i = 0; i < 2; ++i) {
      const int idx = i * 256 + tid;
      const int d = idx & 63, c = (idx >> 6) * 4;
      const size_t base = (size_t)(b * S_ + k0 + c) * H_ + h * DH_ + d;
      *(v4bf*)(Vt + d * 40 + c) =
          cvt4(v[base], v[base + H_], v[base + 2 * H_], v[base + 3 * H_]);
    }
    wait_asynccnt0();
    __syncthreads();

    v16bf af = load_a_frag(Ps, 40, wid * 16, 0);
#pragma unroll
    for (int ni = 0; ni < 4; ++ni) {
      v16bf bf = load_b_frag_t(Vt, 40, 0, ni * 16);
      acc[ni] = wmma_bf16(af, bf, acc[ni]);
    }
  }

  const int lh = lane >> 4;
  const int nn = lane & 15;
#pragma unroll
  for (int ni = 0; ni < 4; ++ni) {
    const int col = h * DH_ + ni * 16 + nn;
#pragma unroll
    for (int r = 0; r < 8; ++r) {
      const int qq = q0 + wid * 16 + r + lh * 8;
      ctx[(size_t)(b * S_ + qq) * H_ + col] = acc[ni][r];
    }
  }
}

// ---------------------------------------------------------------------------
// Block reduction helper (256 threads)
// ---------------------------------------------------------------------------
__device__ inline float block_sum_256(float val, float* red) {
  const int tid = threadIdx.x;
  red[tid] = val;
  __syncthreads();
  for (int s = 128; s > 0; s >>= 1) {
    if (tid < s) red[tid] += red[tid + s];
    __syncthreads();
  }
  const float r = red[0];
  __syncthreads();
  return r;
}

// ---------------------------------------------------------------------------
// Embeddings + importance-noise + LayerNorm. One block per token.
// ---------------------------------------------------------------------------
__global__ __launch_bounds__(256) void embed_ln_kernel(
    float* __restrict__ x, const int* __restrict__ ids,
    const int* __restrict__ tty, const float* __restrict__ imp,
    const float* __restrict__ noise, const float* __restrict__ wemb,
    const float* __restrict__ pemb, const float* __restrict__ temb,
    const float* __restrict__ g, const float* __restrict__ bb) {
  __shared__ float red[256];
  const int t = blockIdx.x;  // token in [0, B*S)
  const int s = t % S_;
  const int tid = threadIdx.x;
  const int wi = ids[t];
  const int ti = tty[t];
  const float im = imp[t];
  const float sigma = 1.0f - im;
  const bool use_noise = (im != 0.0f);

  float e[3];
  float lsum = 0.0f;
#pragma unroll
  for (int j = 0; j < 3; ++j) {
    const int hh = tid + j * 256;
    float vv = wemb[(size_t)wi * H_ + hh] + pemb[(size_t)s * H_ + hh] +
               temb[(size_t)ti * H_ + hh];
    const float nz = noise[(size_t)t * H_ + hh];
    const float nv = vv + nz * sigma * vv;
    vv = use_noise ? nv : vv;
    e[j] = vv;
    lsum += vv;
  }
  const float mean = block_sum_256(lsum, red) * (1.0f / H_);
  float lvar = 0.0f;
#pragma unroll
  for (int j = 0; j < 3; ++j) {
    const float d = e[j] - mean;
    lvar += d * d;
  }
  const float var = block_sum_256(lvar, red) * (1.0f / H_);
  const float rstd = rsqrtf(var + 1e-12f);
#pragma unroll
  for (int j = 0; j < 3; ++j) {
    const int hh = tid + j * 256;
    x[(size_t)t * H_ + hh] = (e[j] - mean) * rstd * g[hh] + bb[hh];
  }
}

// ---------------------------------------------------------------------------
// x = LayerNorm(x + y) * g + b. One block per token. y already includes bias.
// ---------------------------------------------------------------------------
__global__ __launch_bounds__(256) void residual_ln_kernel(
    float* __restrict__ x, const float* __restrict__ y,
    const float* __restrict__ g, const float* __restrict__ bb) {
  __shared__ float red[256];
  const int t = blockIdx.x;
  const int tid = threadIdx.x;
  float e[3];
  float lsum = 0.0f;
#pragma unroll
  for (int j = 0; j < 3; ++j) {
    const size_t idx = (size_t)t * H_ + tid + j * 256;
    e[j] = x[idx] + y[idx];
    lsum += e[j];
  }
  const float mean = block_sum_256(lsum, red) * (1.0f / H_);
  float lvar = 0.0f;
#pragma unroll
  for (int j = 0; j < 3; ++j) {
    const float d = e[j] - mean;
    lvar += d * d;
  }
  const float var = block_sum_256(lvar, red) * (1.0f / H_);
  const float rstd = rsqrtf(var + 1e-12f);
#pragma unroll
  for (int j = 0; j < 3; ++j) {
    const int hh = tid + j * 256;
    x[(size_t)t * H_ + hh] = (e[j] - mean) * rstd * g[hh] + bb[hh];
  }
}

// ---------------------------------------------------------------------------
// Exact GELU in place.
// ---------------------------------------------------------------------------
__global__ __launch_bounds__(256) void gelu_kernel(float* __restrict__ p,
                                                   long long n) {
  const long long stride = (long long)gridDim.x * blockDim.x;
  for (long long i = (long long)blockIdx.x * blockDim.x + threadIdx.x; i < n;
       i += stride) {
    const float u = p[i];
    p[i] = 0.5f * u * (1.0f + erff(u * 0.70710678118654752f));
  }
}

// ---------------------------------------------------------------------------
// Copy x -> d_out
// ---------------------------------------------------------------------------
__global__ __launch_bounds__(256) void copy_kernel(float* __restrict__ dst,
                                                   const float* __restrict__ src,
                                                   long long n) {
  const long long i = (long long)blockIdx.x * blockDim.x + threadIdx.x;
  if (i < n) dst[i] = src[i];
}

// ---------------------------------------------------------------------------
// Pooler: out[b,n] = tanh(x[b,0,:] . pool_w[:,n] + pool_b[n])
// ---------------------------------------------------------------------------
__global__ __launch_bounds__(256) void pooler_kernel(
    float* __restrict__ out, const float* __restrict__ x,
    const float* __restrict__ pw, const float* __restrict__ pb) {
  const int gid = blockIdx.x * 256 + threadIdx.x;
  if (gid >= B_ * H_) return;
  const int b = gid / H_;
  const int n = gid % H_;
  const float* xr = x + (size_t)b * S_ * H_;  // sequence row 0
  float sum = 0.0f;
  for (int kk = 0; kk < H_; ++kk) sum += xr[kk] * pw[(size_t)kk * H_ + n];
  out[gid] = tanhf(sum + pb[n]);
}

// ---------------------------------------------------------------------------
// Host orchestration
// ---------------------------------------------------------------------------
extern "C" void kernel_launch(void* const* d_in, const int* in_sizes, int n_in,
                              void* d_out, int out_size, void* d_ws,
                              size_t ws_size, hipStream_t stream) {
  (void)in_sizes; (void)n_in; (void)out_size; (void)ws_size;
  const int*   input_ids = (const int*)d_in[0];
  const int*   token_ty  = (const int*)d_in[1];
  const float* amask     = (const float*)d_in[2];
  const float* imp       = (const float*)d_in[3];
  const float* noise     = (const float*)d_in[4];
  const float* wemb      = (const float*)d_in[5];
  const float* pemb      = (const float*)d_in[6];
  const float* temb      = (const float*)d_in[7];
  const float* eg        = (const float*)d_in[8];
  const float* ebb       = (const float*)d_in[9];
  const float* Wq = (const float*)d_in[10]; const float* bq = (const float*)d_in[11];
  const float* Wk = (const float*)d_in[12]; const float* bk = (const float*)d_in[13];
  const float* Wv = (const float*)d_in[14]; const float* bv = (const float*)d_in[15];
  const float* Wo = (const float*)d_in[16]; const float* bo = (const float*)d_in[17];
  const float* g1 = (const float*)d_in[18]; const float* b1 = (const float*)d_in[19];
  const float* Wi = (const float*)d_in[20]; const float* bi = (const float*)d_in[21];
  const float* Wd = (const float*)d_in[22]; const float* bd = (const float*)d_in[23];
  const float* g2 = (const float*)d_in[24]; const float* b2 = (const float*)d_in[25];
  const float* pw = (const float*)d_in[26]; const float* pb = (const float*)d_in[27];

  const size_t NX = (size_t)B_ * S_ * H_;  // 6,291,456
  char* ws = (char*)d_ws;
  size_t off = 0;
  auto wsalloc = [&](size_t bytes) -> void* {
    void* p = ws + off;
    off += (bytes + 255) & ~(size_t)255;
    return p;
  };
  float*  x      = (float*)wsalloc(NX * 4);
  float*  q      = (float*)wsalloc(NX * 4);
  float*  k      = (float*)wsalloc(NX * 4);
  float*  v      = (float*)wsalloc(NX * 4);
  float*  ctx    = (float*)wsalloc(NX * 4);
  float*  proj   = (float*)wsalloc(NX * 4);
  float*  ffh    = (float*)wsalloc((size_t)B_ * S_ * FF_ * 4);
  float*  scores = (float*)wsalloc((size_t)B_ * NH_ * S_ * S_ * 4);
  __bf16* probs  = (__bf16*)wsalloc((size_t)B_ * NH_ * S_ * S_ * 2);

  const int M = B_ * S_;  // 8192
  const dim3 blk256(256);

  embed_ln_kernel<<<M, blk256, 0, stream>>>(x, input_ids, token_ty, imp, noise,
                                            wemb, pemb, temb, eg, ebb);

  for (int l = 0; l < L_; ++l) {
    const float* wq_l = Wq + (size_t)l * H_ * H_;
    const float* wk_l = Wk + (size_t)l * H_ * H_;
    const float* wv_l = Wv + (size_t)l * H_ * H_;
    const float* wo_l = Wo + (size_t)l * H_ * H_;
    const float* wi_l = Wi + (size_t)l * H_ * FF_;
    const float* wd_l = Wd + (size_t)l * FF_ * H_;

    gemm_bf16_kernel<<<dim3(M / 128, H_ / 128), blk256, 0, stream>>>(
        q, x, wq_l, bq + (size_t)l * H_, M, H_, H_);
    gemm_bf16_kernel<<<dim3(M / 128, H_ / 128), blk256, 0, stream>>>(
        k, x, wk_l, bk + (size_t)l * H_, M, H_, H_);
    gemm_bf16_kernel<<<dim3(M / 128, H_ / 128), blk256, 0, stream>>>(
        v, x, wv_l, bv + (size_t)l * H_, M, H_, H_);

    attn_scores_kernel<<<dim3(S_ / 128, S_ / 64, B_ * NH_), blk256, 0, stream>>>(
        scores, q, k, amask);
    softmax_kernel<<<(B_ * NH_ * S_) / 4, dim3(128), 0, stream>>>(probs, scores);
    attn_ctx_kernel<<<dim3(S_ / 128, 1, B_ * NH_), blk256, 0, stream>>>(
        ctx, probs, v);

    gemm_bf16_kernel<<<dim3(M / 128, H_ / 128), blk256, 0, stream>>>(
        proj, ctx, wo_l, bo + (size_t)l * H_, M, H_, H_);
    residual_ln_kernel<<<M, blk256, 0, stream>>>(x, proj,
                                                 g1 + (size_t)l * H_,
                                                 b1 + (size_t)l * H_);

    gemm_bf16_kernel<<<dim3(M / 128, FF_ / 128), blk256, 0, stream>>>(
        ffh, x, wi_l, bi + (size_t)l * FF_, M, FF_, H_);
    gelu_kernel<<<4096, blk256, 0, stream>>>(ffh, (long long)M * FF_);
    gemm_bf16_kernel<<<dim3(M / 128, H_ / 128), blk256, 0, stream>>>(
        proj, ffh, wd_l, bd + (size_t)l * H_, M, H_, FF_);
    residual_ln_kernel<<<M, blk256, 0, stream>>>(x, proj,
                                                 g2 + (size_t)l * H_,
                                                 b2 + (size_t)l * H_);
  }

  copy_kernel<<<(int)((NX + 255) / 256), blk256, 0, stream>>>(
      (float*)d_out, x, (long long)NX);
  pooler_kernel<<<(B_ * H_ + 255) / 256, blk256, 0, stream>>>(
      (float*)d_out + NX, x, pw, pb);
}
